// _FPN_64089501990932
// MI455X (gfx1250) — matrix operand order repaired and checked by reading, not compiled, not run on hardware
//
#include <hip/hip_runtime.h>

// ---------------------------------------------------------------------------
// CDNA5 (gfx1250) FPN + ROI-Align, bf16 WMMA (f32 accumulate), wave32.
// conv3x3 inputs are produced in bf16 channel-last so LDS staging is a pure
// byte copy done with GLOBAL_LOAD_ASYNC_TO_LDS_B128 (ASYNCcnt path).
// ---------------------------------------------------------------------------

typedef __attribute__((ext_vector_type(16))) __bf16 v16bf;
typedef __attribute__((ext_vector_type(8)))  float  v8f;

__device__ __forceinline__ unsigned short f2bfbits(float f) {
  unsigned u = __builtin_bit_cast(unsigned, f);
  u += 0x7FFFu + ((u >> 16) & 1u);          // round-to-nearest-even
  return (unsigned short)(u >> 16);
}
__device__ __forceinline__ unsigned pack2bf(float lo, float hi) {
  return (unsigned)f2bfbits(lo) | ((unsigned)f2bfbits(hi) << 16);
}

// ---------------------------------------------------------------------------
// Weight repack: fp32 [Co][Cin][KH][KW] -> bf16 A-fragment tiles.
// A k order chosen here (B reads use the same): element j of the per-lane 16 maps to
//   K = ((lane>>4)<<4) + j,  M = lane&15
// Buffer index: (((tap*mTiles + m)*kTiles + kt)*32 + lane)*16 + j
// ---------------------------------------------------------------------------
__global__ void repack_w(const float* __restrict__ w, __bf16* __restrict__ out,
                         int Co, int Cin, int taps) {
  const int mTiles = Co >> 4, kTiles = Cin >> 5;
  long total = (long)taps * mTiles * kTiles * 512;
  long i = (long)blockIdx.x * blockDim.x + threadIdx.x;
  if (i >= total) return;
  int  j    = (int)(i & 15);
  int  lane = (int)((i >> 4) & 31);
  long tile = i >> 9;
  int  kt = (int)(tile % kTiles); tile /= kTiles;
  int  m  = (int)(tile % mTiles);
  int  t  = (int)(tile / mTiles);
  int  k  = ((lane >> 4) << 4) + j;
  int  co = m * 16 + (lane & 15);
  int  ci = kt * 32 + k;
  float v = w[((long)co * Cin + ci) * taps + t];
  unsigned short hb = f2bfbits(v);
  out[i] = __builtin_bit_cast(__bf16, hb);
}

// ---------------------------------------------------------------------------
// 1x1 conv as GEMM:  y[b,co,p] = bias[co] + sum_ci W[co,ci] * x[b,ci,p]
// Block = 256 threads (8 waves).  Tile: M=128 (8 waves x 16 co), N=32 px.
// Input is fp32 NCHW (harness layout) so staging converts on the fly;
// pixel index is clamped (no exec divergence), edge columns never stored.
// ---------------------------------------------------------------------------
__global__ void __launch_bounds__(256)
conv1x1_wmma(const float* __restrict__ x, const __bf16* __restrict__ wA,
             const float* __restrict__ bias, float* __restrict__ y,
             int Cin, int HW) {
  __shared__ __align__(32) unsigned short xsT[32][32];   // [px][ci]
  const int tid  = threadIdx.x;
  const int lane = tid & 31, wave = tid >> 5;
  const int pixBase = blockIdx.x * 32;
  const int b       = blockIdx.z;
  const int m       = blockIdx.y * 8 + wave;       // co-tile index (of Co/16)
  const int kTiles  = Cin >> 5;
  const long xb = (long)b * Cin * HW;
  v8f acc0 = {}, acc1 = {};

  for (int kt = 0; kt < kTiles; ++kt) {
    __syncthreads();
    #pragma unroll
    for (int u = tid; u < 512; u += 256) {           // stage 32px x 32ci
      int px = u & 31, cp = u >> 5;
      int p  = pixBase + px;
      p = (p < HW) ? p : (HW - 1);                   // clamp: no divergence
      const float* src = x + xb + (long)(kt * 32 + cp * 2) * HW + p;
      *(unsigned*)&xsT[px][cp * 2] = pack2bf(src[0], src[HW]);
    }
    __syncthreads();

    v16bf a = *(const v16bf*)(wA + (((long)m * kTiles + kt) * 32 + lane) * 16);
    const int n  = lane & 15;
    const int kb = (lane >> 4) << 4;                 // K-half base: 0 or 16
    v16bf b0 = *(const v16bf*)&xsT[n][kb];           // 2x ds_load_b128
    v16bf b1 = *(const v16bf*)&xsT[16 + n][kb];
    acc0 = __builtin_amdgcn_wmma_f32_16x16x32_bf16(false, a, false, b0,
                                                   (short)0, acc0, false, false);
    acc1 = __builtin_amdgcn_wmma_f32_16x16x32_bf16(false, a, false, b1,
                                                   (short)0, acc1, false, false);
  }

  const int n = lane & 15;
  const int coBase = m * 16 + ((lane >> 4) << 3);    // C/D layout: M=r+8*lane[4]
  const int p0 = pixBase + n, p1 = pixBase + 16 + n;
  #pragma unroll
  for (int r = 0; r < 8; ++r) {
    int co = coBase + r;
    float bv = bias[co];
    long rowOff = ((long)b * 256 + co) * HW;
    if (p0 < HW) y[rowOff + p0] = acc0[r] + bv;
    if (p1 < HW) y[rowOff + p1] = acc1[r] + bv;
  }
}

// ---------------------------------------------------------------------------
// 3x3 conv (pad=1), Cin=Co=256, implicit GEMM (K = 9*256).
// Input is bf16 channel-last [B][H][W][256], so the LDS halo tile
// xsT[r][xx][ci] is filled by pure 16B async copies:
//   global_load_async_to_lds_b128  (ASYNCcnt), s_wait_asynccnt + barrier.
// Zero padding: LDS is zeroed once; out-of-range units never issue, and
// boundary cells are never overwritten across K-slices.
// Per tap: one A fragment (2x global_load_b128) feeds two WMMAs (N=32).
// ---------------------------------------------------------------------------
__global__ void __launch_bounds__(256)
conv3x3_wmma(const __bf16* __restrict__ xcl, const __bf16* __restrict__ wA,
             const float* __restrict__ bias, float* __restrict__ y,
             int H, int W) {
  __shared__ __align__(32) unsigned short xsT[3][34][32]; // [r][xx][ci]
  const int kTiles = 8, mTiles = 16;
  const int tid  = threadIdx.x;
  const int lane = tid & 31, wave = tid >> 5;
  const int xBase = blockIdx.x * 32;
  const int yRow  = blockIdx.y;
  const int b      = blockIdx.z >> 1;
  const int coHalf = blockIdx.z & 1;
  const int m      = coHalf * 8 + wave;
  const long xb = (long)b * H * W * 256;             // channel-last batch base
  v8f acc0 = {}, acc1 = {};

  // zero the tile once: halo cells outside the image stay zero for all kt
  for (int u = tid; u < 3 * 34 * 4; u += 256) {
    int q = u & 3, xr = u >> 2;
    int xx = xr % 34, r = xr / 34;
    uint4 z = {0u, 0u, 0u, 0u};
    *(uint4*)&xsT[r][xx][q * 8] = z;
  }

  for (int kt = 0; kt < kTiles; ++kt) {
    __syncthreads();                                 // WAR: prior reads done
    // 3 rows x 34 px x 4 chunks of 16B = 408 async units
    for (int u = tid; u < 3 * 34 * 4; u += 256) {
      int q = u & 3, xr = u >> 2;
      int xx = xr % 34, r = xr / 34;
      int gy = yRow + r - 1, gx = xBase + xx - 1;
      if (gy >= 0 && gy < H && gx >= 0 && gx < W) {
        unsigned lds = (unsigned)(unsigned long long)&xsT[r][xx][q * 8];
        unsigned long long g = (unsigned long long)(xcl + xb +
            ((long)(gy * W + gx) * 256 + kt * 32 + q * 8));
        asm volatile("global_load_async_to_lds_b128 %0, %1, off"
                     :: "v"(lds), "v"(g) : "memory");
      }
    }
    asm volatile("s_wait_asynccnt 0x0" ::: "memory");
    __syncthreads();

    const int n  = lane & 15;
    const int kb = (lane >> 4) << 4;
    #pragma unroll
    for (int tap = 0; tap < 9; ++tap) {
      const int ky = tap / 3, kx = tap % 3;
      v16bf a = *(const v16bf*)(wA +
          ((((long)tap * mTiles + m) * kTiles + kt) * 32 + lane) * 16);
      v16bf b0 = *(const v16bf*)&xsT[ky][n + kx][kb];      // 2x ds_load_b128
      v16bf b1 = *(const v16bf*)&xsT[ky][16 + n + kx][kb];
      acc0 = __builtin_amdgcn_wmma_f32_16x16x32_bf16(false, a, false, b0,
                                                     (short)0, acc0, false, false);
      acc1 = __builtin_amdgcn_wmma_f32_16x16x32_bf16(false, a, false, b1,
                                                     (short)0, acc1, false, false);
    }
  }

  const int n = lane & 15;
  const int gx0 = xBase + n, gx1 = xBase + 16 + n;
  const int coBase = m * 16 + ((lane >> 4) << 3);
  #pragma unroll
  for (int r = 0; r < 8; ++r) {
    int co = coBase + r;
    float bv = bias[co];
    long rowOff = (((long)b * 256 + co) * H + yRow) * W;
    if (gx0 < W) y[rowOff + gx0] = acc0[r] + bv;
    if (gx1 < W) y[rowOff + gx1] = acc1[r] + bv;
  }
}

// ---------------------------------------------------------------------------
// outcl[b,y,x,c] = bf16( lat[b,c,y,x] + bilinear_x2(prev)[b,c,y,x] )
// Produces the channel-last bf16 tensor consumed by conv3x3's async staging.
// ---------------------------------------------------------------------------
__global__ void upsample_add_cl(const float* __restrict__ prev,
                                const float* __restrict__ lat,
                                __bf16* __restrict__ outcl,
                                int Hs, int Ws, int H, int W, long total) {
  long i = (long)blockIdx.x * blockDim.x + threadIdx.x;
  if (i >= total) return;
  int x = (int)(i % W); long t = i / W;
  int y = (int)(t % H); long t2 = t / H;
  int c = (int)(t2 % 256);
  int b = (int)(t2 / 256);
  float sy = (y + 0.5f) * 0.5f - 0.5f;
  float sx = (x + 0.5f) * 0.5f - 0.5f;
  sy = fminf(fmaxf(sy, 0.0f), (float)Hs - 1.0f);
  sx = fminf(fmaxf(sx, 0.0f), (float)Ws - 1.0f);
  int y0 = (int)sy, x0 = (int)sx;
  int y1 = (y0 + 1 < Hs) ? y0 + 1 : Hs - 1;
  int x1 = (x0 + 1 < Ws) ? x0 + 1 : Ws - 1;
  float ly = sy - y0, lx = sx - x0;
  const float* p = prev + ((long)b * 256 + c) * Hs * Ws;
  float v = (1.0f - ly) * (1.0f - lx) * p[y0 * Ws + x0]
          + (1.0f - ly) * lx          * p[y0 * Ws + x1]
          + ly          * (1.0f - lx) * p[y1 * Ws + x0]
          + ly          * lx          * p[y1 * Ws + x1];
  v += lat[i];
  unsigned short hb = f2bfbits(v);
  outcl[(((long)b * H + y) * W + x) * 256 + c] = __builtin_bit_cast(__bf16, hb);
}

// ---------------------------------------------------------------------------
// ROI-Align with FPN level assignment (P=7, SAMP=2). One lane per output.
// ---------------------------------------------------------------------------
__global__ void roi_align_kernel(const float* __restrict__ rois,
                                 const float* __restrict__ im_info,
                                 const float* __restrict__ p2,
                                 const float* __restrict__ p3,
                                 const float* __restrict__ p4,
                                 const float* __restrict__ p5,
                                 float* __restrict__ out, int N) {
  long i = (long)blockIdx.x * blockDim.x + threadIdx.x;
  long total = (long)N * 256 * 49;
  if (i >= total) return;
  int pp = (int)(i % 49); long t = i / 49;
  int c  = (int)(t % 256);
  int n  = (int)(t / 256);
  int py = pp / 7, px = pp % 7;

  const float* roi = rois + (long)n * 5;
  float rh = roi[4] - roi[2] + 1.0f;
  float rw = roi[3] - roi[1] + 1.0f;
  float lvl = rintf(logf(sqrtf(rh * rw) / 224.0f) + 4.0f);
  lvl = fminf(fmaxf(lvl, 2.0f), 5.0f);
  int li = (int)lvl - 2;

  const float* f = (li == 0) ? p2 : (li == 1) ? p3 : (li == 2) ? p4 : p5;
  int FH = (li == 0) ? 200 : (li == 1) ? 100 : (li == 2) ? 50 : 25;
  int FW = (li == 0) ? 304 : (li == 1) ? 152 : (li == 2) ? 76 : 38;
  float scale = (float)FH / im_info[0];
  int b = (int)roi[0];
  const float* img = f + ((long)b * 256 + c) * (long)FH * FW;

  float x1 = roi[1] * scale, y1 = roi[2] * scale;
  float x2 = roi[3] * scale, y2 = roi[4] * scale;
  float w_ = fmaxf(x2 - x1, 1.0f), h_ = fmaxf(y2 - y1, 1.0f);
  float bh = h_ / 7.0f, bw = w_ / 7.0f;

  float accv = 0.0f;
  #pragma unroll
  for (int sy = 0; sy < 2; ++sy) {
    #pragma unroll
    for (int sx = 0; sx < 2; ++sx) {
      float ys = y1 + py * bh + (sy + 0.5f) * bh * 0.5f;
      float xsv = x1 + px * bw + (sx + 0.5f) * bw * 0.5f;
      bool valid = (ys >= -1.0f) && (ys <= (float)FH) &&
                   (xsv >= -1.0f) && (xsv <= (float)FW);
      float yc = fminf(fmaxf(ys, 0.0f), (float)FH - 1.0f);
      float xc = fminf(fmaxf(xsv, 0.0f), (float)FW - 1.0f);
      int y0 = (int)floorf(yc), x0 = (int)floorf(xc);
      int y1i = (y0 + 1 < FH) ? y0 + 1 : FH - 1;
      int x1i = (x0 + 1 < FW) ? x0 + 1 : FW - 1;
      float ly = yc - (float)y0, lx = xc - (float)x0;
      float hy = 1.0f - ly, hx = 1.0f - lx;
      float v = hy * hx * img[y0 * FW + x0] + hy * lx * img[y0 * FW + x1i]
              + ly * hx * img[y1i * FW + x0] + ly * lx * img[y1i * FW + x1i];
      accv += valid ? v : 0.0f;
    }
  }
  out[i] = accv * 0.25f;
}

// ---------------------------------------------------------------------------
// Host-side launch
// ---------------------------------------------------------------------------
extern "C" void kernel_launch(void* const* d_in, const int* in_sizes, int n_in,
                              void* d_out, int out_size, void* d_ws, size_t ws_size,
                              hipStream_t stream) {
  (void)in_sizes; (void)n_in; (void)out_size; (void)ws_size;
  const float* c2      = (const float*)d_in[0];
  const float* c3      = (const float*)d_in[1];
  const float* c4      = (const float*)d_in[2];
  const float* c5      = (const float*)d_in[3];
  const float* rois    = (const float*)d_in[4];
  const float* im_info = (const float*)d_in[5];
  const float* w_top  = (const float*)d_in[6];  const float* b_top  = (const float*)d_in[7];
  const float* w_lat1 = (const float*)d_in[8];  const float* b_lat1 = (const float*)d_in[9];
  const float* w_lat2 = (const float*)d_in[10]; const float* b_lat2 = (const float*)d_in[11];
  const float* w_lat3 = (const float*)d_in[12]; const float* b_lat3 = (const float*)d_in[13];
  const float* w_sm1  = (const float*)d_in[14]; const float* b_sm1  = (const float*)d_in[15];
  const float* w_sm2  = (const float*)d_in[16]; const float* b_sm2  = (const float*)d_in[17];
  const float* w_sm3  = (const float*)d_in[18]; const float* b_sm3  = (const float*)d_in[19];
  float* out = (float*)d_out;

  const int B = 2, N = 1024;
  const long HW5 = 25L * 38, HW4 = 50L * 76, HW3 = 100L * 152, HW2 = 200L * 304;

  // ---- carve workspace ----
  char* ws = (char*)d_ws;
  size_t off = 0;
  auto carve = [&](size_t bytes) -> void* {
    void* p = ws + off;
    off += (bytes + 255) & ~(size_t)255;
    return p;
  };
  float* p5    = (float*)carve((size_t)B * 256 * HW5 * 4);
  float* lat4  = (float*)carve((size_t)B * 256 * HW4 * 4);
  float* p4    = (float*)carve((size_t)B * 256 * HW4 * 4);
  float* lat3  = (float*)carve((size_t)B * 256 * HW3 * 4);
  float* p3    = (float*)carve((size_t)B * 256 * HW3 * 4);
  float* lat2  = (float*)carve((size_t)B * 256 * HW2 * 4);
  float* p2    = (float*)carve((size_t)B * 256 * HW2 * 4);
  __bf16* t4bf = (__bf16*)carve((size_t)B * 256 * HW4 * 2);  // channel-last
  __bf16* t3bf = (__bf16*)carve((size_t)B * 256 * HW3 * 2);
  __bf16* t2bf = (__bf16*)carve((size_t)B * 256 * HW2 * 2);
  __bf16* wTopA  = (__bf16*)carve(1L * 16 * 64 * 512 * 2);  // taps*mTiles*kTiles*512
  __bf16* wLat1A = (__bf16*)carve(1L * 16 * 32 * 512 * 2);
  __bf16* wLat2A = (__bf16*)carve(1L * 16 * 16 * 512 * 2);
  __bf16* wLat3A = (__bf16*)carve(1L * 16 * 8  * 512 * 2);
  __bf16* wSm1A  = (__bf16*)carve(9L * 16 * 8  * 512 * 2);
  __bf16* wSm2A  = (__bf16*)carve(9L * 16 * 8  * 512 * 2);
  __bf16* wSm3A  = (__bf16*)carve(9L * 16 * 8  * 512 * 2);

  // ---- weight repack (bf16, A-fragment order) ----
  auto repack = [&](const float* w, __bf16* o, int Co, int Cin, int taps) {
    long total = (long)taps * (Co >> 4) * (Cin >> 5) * 512;
    repack_w<<<(unsigned)((total + 255) / 256), 256, 0, stream>>>(w, o, Co, Cin, taps);
  };
  repack(w_top,  wTopA,  256, 2048, 1);
  repack(w_lat1, wLat1A, 256, 1024, 1);
  repack(w_lat2, wLat2A, 256, 512,  1);
  repack(w_lat3, wLat3A, 256, 256,  1);
  repack(w_sm1,  wSm1A,  256, 256,  9);
  repack(w_sm2,  wSm2A,  256, 256,  9);
  repack(w_sm3,  wSm3A,  256, 256,  9);

  auto c1 = [&](const float* x, const __bf16* wA, const float* bias, float* y,
                int Cin, long HW) {
    dim3 g((unsigned)((HW + 31) / 32), 2, B);
    conv1x1_wmma<<<g, 256, 0, stream>>>(x, wA, bias, y, Cin, (int)HW);
  };
  auto c3k = [&](const __bf16* xcl, const __bf16* wA, const float* bias, float* y,
                 int H, int W) {
    dim3 g((unsigned)((W + 31) / 32), (unsigned)H, B * 2);
    conv3x3_wmma<<<g, 256, 0, stream>>>(xcl, wA, bias, y, H, W);
  };
  auto up = [&](const float* prev, const float* lat, __bf16* outcl,
                int Hs, int Ws, int H, int W) {
    long total = (long)B * 256 * H * W;
    upsample_add_cl<<<(unsigned)((total + 255) / 256), 256, 0, stream>>>(
        prev, lat, outcl, Hs, Ws, H, W, total);
  };

  // ---- FPN top-down ----
  c1(c5, wTopA, b_top, p5, 2048, HW5);                 // p5
  c1(c4, wLat1A, b_lat1, lat4, 1024, HW4);             // lateral 4
  up(p5, lat4, t4bf, 25, 38, 50, 76);                  // t4 = up(p5)+lat4 (bf16 CL)
  c3k(t4bf, wSm1A, b_sm1, p4, 50, 76);                 // p4
  c1(c3, wLat2A, b_lat2, lat3, 512, HW3);              // lateral 3
  up(p4, lat3, t3bf, 50, 76, 100, 152);
  c3k(t3bf, wSm2A, b_sm2, p3, 100, 152);               // p3
  c1(c2, wLat3A, b_lat3, lat2, 256, HW2);              // lateral 2
  up(p3, lat2, t2bf, 100, 152, 200, 304);
  c3k(t2bf, wSm3A, b_sm3, p2, 200, 304);               // p2

  // ---- ROI-Align ----
  long total = (long)N * 256 * 49;
  roi_align_kernel<<<(unsigned)((total + 255) / 256), 256, 0, stream>>>(
      rois, im_info, p2, p3, p4, p5, out, N);
}
